// ThalamusModel_73435350827613
// MI455X (gfx1250) — matrix operand order repaired and checked by reading, not compiled, not run on hardware
//
#include <hip/hip_runtime.h>
#include <hip/hip_bf16.h>

typedef __attribute__((ext_vector_type(16))) _Float16 v16h;
typedef __attribute__((ext_vector_type(8)))  float    v8f;
typedef __attribute__((ext_vector_type(4)))  float    v4f;

#define B_   64
#define S_   256
#define DM_  66
#define DI_  132
#define WS_  64
#define DCONV_ 4
#define DTR_ 5

// ---------------------------------------------------------------------------
// Device helpers
// ---------------------------------------------------------------------------
__device__ __forceinline__ float sigmoidf_(float x) { return 1.f / (1.f + __expf(-x)); }
__device__ __forceinline__ float siluf_(float x)    { return x * sigmoidf_(x); }
__device__ __forceinline__ float softplusf_(float x){ return (x > 20.f) ? x : log1pf(__expf(x)); }

// ---------------------------------------------------------------------------
// Weight pack: f32 row-major (M x K) -> f16 WMMA A-fragment tiles.
// Tile (mt, kt): 32 lanes x 16 halves; lane l (row = mt*16 + (l&15)) holds
// K = kt*32 + ((l&16)?16:0) + e  for e = 0..15  (contiguous k order; B loads match).
// Padding rows/cols are zero so padded outputs compute to exactly 0.
// ---------------------------------------------------------------------------
__global__ void pack_w_kernel(const float* __restrict__ W, _Float16* __restrict__ dst,
                              int M, int K, int ktiles) {
  int tile = blockIdx.x;
  int mt = tile / ktiles, kt = tile % ktiles;
  for (int j = threadIdx.x; j < 512; j += blockDim.x) {
    int l = j >> 4, e = j & 15;
    int row = mt * 16 + (l & 15);
    int k   = kt * 32 + ((l & 16) ? 16 : 0) + e;
    float v = (row < M && k < K) ? W[(size_t)row * K + k] : 0.f;
    dst[(size_t)tile * 512 + j] = (_Float16)v;
  }
}

__global__ void aneg_kernel(const float* __restrict__ A_log, float* __restrict__ Aneg, int n) {
  int i = blockIdx.x * blockDim.x + threadIdx.x;
  if (i < n) Aneg[i] = -__expf(A_log[i]);
}

// ---------------------------------------------------------------------------
// Cooperative mat-vec via v_wmma_f32_16x16x32_f16 (compile-time tile counts).
// out[0..MT*16) (f32 LDS, padded to multiple of 16, 16B aligned)
//   = Wt (packed f16 tiles) * uh (f16 LDS, zero padded to KT*32).
// B fragment: u replicated into all 16 columns (same contiguous k order as A: lanes0-15 K=0..15,
// lanes16-31 K=16..31) -> one aligned 32B ds load per k-tile.
// D extraction: column N==0 lives in lanes 0 (rows mt*16+0..7) and 16
// (rows mt*16+8..15); store the v8f accumulator as two ds_store_b128.
// Must be called from uniform control flow (EXEC all ones for WMMA).
// ---------------------------------------------------------------------------
template <int KT, int MT>
__device__ __forceinline__ void wmma_matvec(const _Float16* __restrict__ Wt,
                                            const _Float16* __restrict__ uh,
                                            float* __restrict__ out) {
  const int lane = threadIdx.x & 31;
  const int wave = threadIdx.x >> 5;

  v16h bf[KT];
#pragma unroll
  for (int kt = 0; kt < KT; ++kt) {
    int kb = kt * 32 + ((lane & 16) ? 16 : 0);
    bf[kt] = *(const v16h*)(uh + kb);            // 32B-aligned ds_load
  }
#pragma unroll
  for (int mt = wave; mt < MT; mt += 8) {        // blockDim = 256 -> 8 waves
    v8f acc = {};
#pragma unroll
    for (int kt = 0; kt < KT; ++kt) {
      const v16h a = *(const v16h*)(Wt + ((size_t)(mt * KT + kt) * 512) + lane * 16);
      acc = __builtin_amdgcn_wmma_f32_16x16x32_f16(
          /*neg_a=*/false, a, /*neg_b=*/false, bf[kt],
          /*c_mod=*/(short)0, acc, /*reuse_a=*/false, /*reuse_b=*/false);
    }
    if ((lane & 15) == 0) {
      int mb = mt * 16 + ((lane & 16) ? 8 : 0);
      v4f lo = __builtin_shufflevector(acc, acc, 0, 1, 2, 3);
      v4f hi = __builtin_shufflevector(acc, acc, 4, 5, 6, 7);
      *(v4f*)(out + mb)     = lo;                // ds_store_b128
      *(v4f*)(out + mb + 4) = hi;                // ds_store_b128
    }
  }
}

// ---------------------------------------------------------------------------
// Parameter block
// ---------------------------------------------------------------------------
struct BlockP {
  const _Float16 *in_p, *x_p, *dt_p, *out_p;   // packed f16 tiles in d_ws
  const float *conv_w, *conv_b, *dt_b, *Aneg, *D;
  int ds;
};
struct WsP {
  const _Float16* proj_p;                      // 3 contiguous packs of 6144 halfs
  const float *proj_b, *gate_w, *gate_b, *ln_g, *ln_b, *head_w, *head_b;
};
struct Params {
  const float* x;
  BlockP blk[3];
  WsP ws;
  float* out;      // (B,S,1)
  float* gates;    // (S,B,3,1)
};

struct SharedBufs {
  _Float16* uh;
  float *cv, *u, *xz, *xc, *xdb, *dtv, *yv, *eo;
};

// LDS float budget after the 160-half (80-float) f16 staging buffer.
// Every chunk is a multiple of 4 floats so all buffers stay 16B aligned.
#define SMEM_FLOATS_ (27456 + 1584 + 68 + 272 + 132 + 272 + 144 + 132 + 240 + 192 + 64 + 64 + 4 + 4 + 8)
#define SMEM_BYTES_  ((80 + SMEM_FLOATS_) * 4)

// ---------------------------------------------------------------------------
// One fully-specialized mamba block step (DS = 16/64/128 compile-time).
// ---------------------------------------------------------------------------
template <int DS, int IDX, int SSMOFF>
__device__ __forceinline__ void mamba_block(const BlockP& Q, float* __restrict__ ssm,
                                            const SharedBufs& S, int t) {
  const int tid = threadIdx.x, lane = tid & 31, wave = tid >> 5;

  // in_proj: xz(264) = W(264x66) * u
  if (tid < 96) S.uh[tid] = (tid < DM_) ? (_Float16)S.u[tid] : (_Float16)0.f;
  __syncthreads();
  wmma_matvec<3, 17>(Q.in_p, S.uh, S.xz);
  __syncthreads();

  // depthwise conv (ring buffer) + SiLU
  if (tid < DI_) {
    int d = tid;
    float* cb = S.cv + (IDX * DI_ + d) * DCONV_;
    int p = t & 3;
    cb[p] = S.xz[d];
    float s = Q.conv_b[d];
#pragma unroll
    for (int j = 0; j < 4; ++j) s += Q.conv_w[d * 4 + j] * cb[(p + 1 + j) & 3];
    S.xc[d] = siluf_(s);
  }
  __syncthreads();

  // x_proj: xdb(5+2DS) = W * xc
  if (tid < 160) S.uh[tid] = (tid < DI_) ? (_Float16)S.xc[tid] : (_Float16)0.f;
  __syncthreads();
  constexpr int XM = DTR_ + 2 * DS;
  wmma_matvec<5, (XM + 15) / 16>(Q.x_p, S.uh, S.xdb);
  __syncthreads();

  // dt_proj + softplus
  if (tid < 32) S.uh[tid] = (tid < DTR_) ? (_Float16)S.xdb[tid] : (_Float16)0.f;
  __syncthreads();
  wmma_matvec<1, 9>(Q.dt_p, S.uh, S.dtv);
  __syncthreads();
  if (tid < DI_) S.dtv[tid] = softplusf_(S.dtv[tid] + Q.dt_b[tid]);
  __syncthreads();

  // SSM state update + y = ssm @ C + D*xc  (waves over rows, lanes over n;
  // fully unrolled n loop: 1/2/4 iterations for DS = 16/64/128)
  const float* Bm = S.xdb + DTR_;
  const float* Cm = S.xdb + DTR_ + DS;
  float* st = ssm + SSMOFF;
  for (int d = wave; d < DI_; d += 8) {
    float dt_d = S.dtv[d], xc_d = S.xc[d];
    float dtxc = dt_d * xc_d;
    float part = 0.f;
#pragma unroll
    for (int nn = 0; nn < DS; nn += 32) {
      int n = nn + lane;
      if (DS >= 32 || lane < DS) {
        float dA = __expf(dt_d * Q.Aneg[d * DS + n]);
        float sN = st[d * DS + n] * dA + dtxc * Bm[n];
        st[d * DS + n] = sN;
        part += sN * Cm[n];
      }
    }
#pragma unroll
    for (int off = 16; off > 0; off >>= 1) part += __shfl_xor(part, off, 32);
    if (lane == 0) S.yv[d] = part + Q.D[d] * xc_d;
  }
  __syncthreads();
  if (tid < DI_) S.yv[tid] *= siluf_(S.xz[DI_ + tid]);   // gate by silu(z)
  __syncthreads();

  // out_proj: eo_IDX(66) = W(66x132) * y
  if (tid < 160) S.uh[tid] = (tid < DI_) ? (_Float16)S.yv[tid] : (_Float16)0.f;
  __syncthreads();
  wmma_matvec<5, 5>(Q.out_p, S.uh, S.eo + IDX * 80);
  __syncthreads();
}

// ---------------------------------------------------------------------------
// Persistent scan kernel: one workgroup per batch element; all recurrent
// state lives in LDS for the entire S=256 scan.
// ---------------------------------------------------------------------------
__global__ __launch_bounds__(256) void thalamus_scan_kernel(Params P) {
  extern __shared__ unsigned char smraw[];
  SharedBufs S;
  S.uh = (_Float16*)smraw;                     // 160 halfs, 32B aligned
  float* f   = (float*)smraw + 80;
  float* ssm = f; f += DI_ * (16 + 64 + 128);  // 27456
  S.cv  = f; f += 3 * DI_ * DCONV_;            // 1584
  S.u   = f; f += 68;                          // 66 used
  S.xz  = f; f += 272;                         // 264 used (xb = xz, z = xz+132)
  S.xc  = f; f += 132;
  S.xdb = f; f += 272;                         // up to 261 used + pad
  S.dtv = f; f += 144;                         // 132 used
  S.yv  = f; f += 132;
  S.eo  = f; f += 3 * 80;                      // 66 used per block, stride 80
  float* pv  = f; f += 3 * WS_;
  float* wsv = f; f += WS_;
  float* gc  = f; f += WS_;
  float* fat = f; f += 4;
  float* wts = f; f += 4;
  float* red = f; f += 8;

  const int tid = threadIdx.x, lane = tid & 31, wave = tid >> 5;
  const int b = blockIdx.x;

  for (int i = tid; i < DI_ * (16 + 64 + 128); i += blockDim.x) ssm[i] = 0.f;
  for (int i = tid; i < 3 * DI_ * DCONV_; i += blockDim.x) S.cv[i] = 0.f;
  if (tid < WS_) wsv[tid] = 0.f;
  if (tid < 3)  fat[tid] = 0.f;
  __syncthreads();

  for (int t = 0; t < S_; ++t) {
    // combined = concat(x_t, ws)
    if (tid < 2)   S.u[tid]     = P.x[((size_t)b * S_ + t) * 2 + tid];
    if (tid < WS_) S.u[2 + tid] = wsv[tid];
    __syncthreads();

    // ---- three mamba blocks (fully specialized) -----------------------
    mamba_block<16,  0, 0>              (P.blk[0], ssm, S, t);
    mamba_block<64,  1, DI_ * 16>       (P.blk[1], ssm, S, t);
    mamba_block<128, 2, DI_ * (16 + 64)>(P.blk[2], ssm, S, t);

    // ---- working-set gating layer -------------------------------------
#pragma unroll
    for (int k = 0; k < 3; ++k) {
      if (tid < 96)
        S.uh[tid] = (tid < DM_) ? (_Float16)(S.eo[k * 80 + tid] * (1.f - fat[k]))
                                : (_Float16)0.f;
      __syncthreads();
      wmma_matvec<3, 4>(P.ws.proj_p + (size_t)k * 6144, S.uh, pv + k * WS_);
      __syncthreads();
    }
    if (tid < 3 * WS_) pv[tid] += P.ws.proj_b[tid];
    __syncthreads();

    // gate scores: 3 waves, lane-parallel 64-dots + shfl reduction
    if (wave < 3) {
      float s = pv[wave * WS_ + lane]      * P.ws.gate_w[lane]
              + pv[wave * WS_ + lane + 32] * P.ws.gate_w[lane + 32];
#pragma unroll
      for (int off = 16; off > 0; off >>= 1) s += __shfl_xor(s, off, 32);
      if (lane == 0) wts[wave] = s + P.ws.gate_b[0];
    }
    __syncthreads();
    if (tid == 0) {           // softmax(scores / 0.05)
      float m = fmaxf(wts[0], fmaxf(wts[1], wts[2]));
      float e0 = __expf((wts[0] - m) * 20.f);
      float e1 = __expf((wts[1] - m) * 20.f);
      float e2 = __expf((wts[2] - m) * 20.f);
      float inv = 1.f / (e0 + e1 + e2);
      wts[0] = e0 * inv; wts[1] = e1 * inv; wts[2] = e2 * inv;
    }
    __syncthreads();
    if (tid < 3) {
      float w = wts[tid];
      P.gates[((size_t)t * B_ + b) * 3 + tid] = w;
      float nf = (fat[tid] + w * 0.15f) * 0.85f;
      fat[tid] = fminf(fmaxf(nf, 0.f), 0.6f);
    }
    if (tid < WS_)
      gc[tid] = pv[tid] * wts[0] + pv[WS_ + tid] * wts[1] + pv[2 * WS_ + tid] * wts[2];
    __syncthreads();

    if (wave == 0) {          // LayerNorm stats over 64 elems
      float v0 = gc[lane], v1 = gc[lane + 32];
      float s = v0 + v1, q = v0 * v0 + v1 * v1;
#pragma unroll
      for (int off = 16; off > 0; off >>= 1) {
        s += __shfl_xor(s, off, 32);
        q += __shfl_xor(q, off, 32);
      }
      if (lane == 0) {
        float mu = s * (1.f / 64.f);
        float var = q * (1.f / 64.f) - mu * mu;
        red[0] = mu;
        red[1] = rsqrtf(var + 1e-5f);
      }
    }
    __syncthreads();
    if (tid < WS_)
      wsv[tid] = (gc[tid] - red[0]) * red[1] * P.ws.ln_g[tid] + P.ws.ln_b[tid];
    __syncthreads();
    if (wave == 0) {          // head: scalar out
      float s = wsv[lane] * P.ws.head_w[lane] + wsv[lane + 32] * P.ws.head_w[lane + 32];
#pragma unroll
      for (int off = 16; off > 0; off >>= 1) s += __shfl_xor(s, off, 32);
      if (lane == 0) P.out[(size_t)b * S_ + t] = s + P.ws.head_b[0];
    }
    __syncthreads();
  }
}

// ---------------------------------------------------------------------------
// Host launcher
// ---------------------------------------------------------------------------
extern "C" void kernel_launch(void* const* d_in, const int* in_sizes, int n_in,
                              void* d_out, int out_size, void* d_ws, size_t ws_size,
                              hipStream_t stream) {
  (void)in_sizes; (void)n_in; (void)out_size; (void)ws_size;

  Params P;
  P.x = (const float*)d_in[0];
  P.out = (float*)d_out;
  P.gates = (float*)d_out + (size_t)B_ * S_;

  _Float16* wsbase = (_Float16*)d_ws;
  size_t off = 0;   // in halfs
  auto packed = [&](const float* W, int M, int K) -> const _Float16* {
    int mt = (M + 15) / 16, kt = (K + 31) / 32;
    _Float16* dst = wsbase + off;
    off += (size_t)mt * kt * 512;
    pack_w_kernel<<<dim3(mt * kt), dim3(256), 0, stream>>>(W, dst, M, K, kt);
    return dst;
  };

  const int dss[3] = {16, 64, 128};
  const float* alog[3];
  for (int i = 0; i < 3; ++i) {
    int base = 1 + i * 9;
    const float* inp   = (const float*)d_in[base + 0];
    const float* convw = (const float*)d_in[base + 1];
    const float* convb = (const float*)d_in[base + 2];
    const float* xp    = (const float*)d_in[base + 3];
    const float* dtp   = (const float*)d_in[base + 4];
    const float* dtb   = (const float*)d_in[base + 5];
    alog[i]            = (const float*)d_in[base + 6];
    const float* Dp    = (const float*)d_in[base + 7];
    const float* outp  = (const float*)d_in[base + 8];

    P.blk[i].in_p  = packed(inp, 2 * DI_, DM_);            // 264 x 66
    P.blk[i].x_p   = packed(xp, DTR_ + 2 * dss[i], DI_);   // (5+2ds) x 132
    P.blk[i].dt_p  = packed(dtp, DI_, DTR_);               // 132 x 5
    P.blk[i].out_p = packed(outp, DM_, DI_);               // 66 x 132
    P.blk[i].conv_w = convw; P.blk[i].conv_b = convb;
    P.blk[i].dt_b = dtb; P.blk[i].D = Dp; P.blk[i].ds = dss[i];
  }

  const float* projw = (const float*)d_in[28];
  const _Float16* proj_p0 = nullptr;
  for (int k = 0; k < 3; ++k) {
    const _Float16* p = packed(projw + (size_t)k * WS_ * DM_, WS_, DM_);  // 64 x 66
    if (k == 0) proj_p0 = p;
  }
  P.ws.proj_p = proj_p0;
  P.ws.proj_b = (const float*)d_in[29];
  P.ws.gate_w = (const float*)d_in[30];
  P.ws.gate_b = (const float*)d_in[31];
  P.ws.ln_g   = (const float*)d_in[32];
  P.ws.ln_b   = (const float*)d_in[33];
  P.ws.head_w = (const float*)d_in[34];
  P.ws.head_b = (const float*)d_in[35];

  // Precompute A = -exp(A_log) into scratch (L2-resident, re-read every step)
  off = (off + 1) & ~(size_t)1;                    // 4B align
  float* anegbase = (float*)(wsbase + off);
  size_t aoff = 0;
  for (int i = 0; i < 3; ++i) {
    int n = DI_ * dss[i];
    aneg_kernel<<<dim3((n + 255) / 256), dim3(256), 0, stream>>>(alog[i], anegbase + aoff, n);
    P.blk[i].Aneg = anegbase + aoff;
    aoff += n;
  }

  thalamus_scan_kernel<<<dim3(B_), dim3(256), SMEM_BYTES_, stream>>>(P);
}
